// LBP_57492432224380
// MI455X (gfx1250) — compile-verified, matrix operand order checked by reading, not run on hardware
//
#include <hip/hip_runtime.h>

// ---------------------------------------------------------------------------
// CDNA5 (gfx1250) implementation.
//  K0: one-time f32->f16 weight conversion (w1, w2) -> L2-resident f16 weights
//  K1: conv1 3x3 (512->256) implicit-GEMM WMMA, TILE_M=128 x N=256, BN+leaky
//  K2: conv2 1x1 (256->128) WMMA GEMM, A-row fragments register-resident
//  K3: x_ave channel sums (deterministic reduction)
//  K4: cosine map
//  K5: per-image LBP histogram + MLP + attention, all GEMMs via WMMA
//  K6: bilinear resize 16x16 -> 48x48
// ---------------------------------------------------------------------------

typedef __attribute__((ext_vector_type(16))) _Float16 v16h;
typedef __attribute__((ext_vector_type(8)))  float    v8f;

// ---------------- workspace layout (bytes) ----------------
static constexpr size_t NPIX      = (size_t)32 * 48 * 48;      // 73728 pixels
static constexpr size_t X2_OFF    = 0;                          // f16 [pix][128]
static constexpr size_t X2_BYTES  = NPIX * 128 * 2;
static constexpr size_t XAVE_OFF  = X2_OFF + X2_BYTES;          // f32 [32][128]
static constexpr size_t XAVE_BYTES= 32 * 128 * 4;
static constexpr size_t COS_OFF   = XAVE_OFF + XAVE_BYTES;      // f32 [32][2304]
static constexpr size_t COS_BYTES = (size_t)32 * 2304 * 4;
static constexpr size_t QUANT_OFF = COS_OFF + COS_BYTES;        // f32 [32][256][128]
static constexpr size_t QUANT_BYTES = (size_t)32 * 256 * 128 * 4;
static constexpr size_t OUTS_OFF  = QUANT_OFF + QUANT_BYTES;    // f32 [32][256][256]
static constexpr size_t OUTS_BYTES= (size_t)32 * 256 * 256 * 4;
static constexpr size_t IMG_OFF   = OUTS_OFF + OUTS_BYTES;      // per-image scratch
// per-image scratch offsets (f16 matrices, attn scores f32)
static constexpr size_t OFF_H1  = 0;                         // 128x64  f16
static constexpr size_t OFF_S   = OFF_H1  + 128*64*2;        // 128x256 f16 (h2 | xa)
static constexpr size_t OFF_S2  = OFF_S   + 128*256*2;       // 128x256 f16
static constexpr size_t OFF_K   = OFF_S2  + 128*256*2;       // 128x256 f16
static constexpr size_t OFF_Q   = OFF_K   + 128*256*2;       // 128x256 f16
static constexpr size_t OFF_V   = OFF_Q   + 128*256*2;       // 256x128 f16 (channel-major)
static constexpr size_t OFF_AF  = OFF_V   + 256*128*2;       // 128x128 f32
static constexpr size_t OFF_A16 = OFF_AF  + 128*128*4;       // 128x128 f16
static constexpr size_t OFF_F   = OFF_A16 + 128*128*2;       // 128x256 f16
static constexpr size_t OFF_F2  = OFF_F   + 128*256*2;       // 256x128 f16 (channel-major)
static constexpr size_t IMG_STRIDE = OFF_F2 + 256*128*2;     // 573440 B per image
// f16 weight copies + conv1 output
static constexpr size_t W1H_OFF   = IMG_OFF + 32 * IMG_STRIDE; // f16 [256][4608]
static constexpr size_t W1H_BYTES = (size_t)256 * 4608 * 2;
static constexpr size_t W2H_OFF   = W1H_OFF + W1H_BYTES;       // f16 [128][256]
static constexpr size_t W2H_BYTES = (size_t)128 * 256 * 2;
static constexpr size_t Y_OFF     = W2H_OFF + W2H_BYTES;       // f16 [pix][256]
static constexpr size_t Y_BYTES   = NPIX * 256 * 2;            // 37.7 MB, L2-resident

// ---------------- WMMA fragment helpers ----------------
// 16-bit A/B operand layout (ISA 7.12.2): lane<16 holds K {0..7,16..23},
// lane>=16 holds K {8..15,24..31}; caller applies koff = (lane>=16) ? 8 : 0.
__device__ __forceinline__ v16h frag_f16(const _Float16* p) {
  v16h x;
#pragma unroll
  for (int i = 0; i < 8; ++i) { x[i] = p[i]; x[i + 8] = p[i + 16]; }
  return x;
}
__device__ __forceinline__ v16h frag_f32(const float* p) {
  v16h x;
#pragma unroll
  for (int i = 0; i < 8; ++i) { x[i] = (_Float16)p[i]; x[i + 8] = (_Float16)p[i + 16]; }
  return x;
}
__device__ __forceinline__ v8f wmma32(v16h a, v16h b, v8f c) {
  return __builtin_amdgcn_wmma_f32_16x16x32_f16(false, a, false, b, (short)0, c,
                                                false, false);
}

// ---------------- generic block GEMM (operands in global memory) ----------
// C(MxN) = A(MxK) * B(NxK)^T ; row-major, K multiple of 32, M/N multiples of 16.
// EPI: 0 none, 1 BN(col)+relu, 2 BN(row)+relu.
template <bool AF16, bool BF16, int EPI, bool OUTF16>
__device__ void gemm_block(const void* __restrict__ A, int lda,
                           const void* __restrict__ B, int ldb,
                           void* __restrict__ C, int ldc,
                           int M, int N, int K,
                           const float* g, const float* bb,
                           const float* mm, const float* vv) {
  const int tid  = threadIdx.x;
  const int lane = tid & 31;
  const int wave = tid >> 5;
  const int nw   = blockDim.x >> 5;
  const int hl   = lane >> 4;          // half-wave selector
  const int lr   = lane & 15;
  const int koff = hl ? 8 : 0;
  const int mt = M >> 4, nt = N >> 4;
  for (int t = wave; t < mt * nt; t += nw) {
    const int mi = t / nt, ni = t - mi * nt;
    v8f acc = {};
    for (int k0 = 0; k0 < K; k0 += 32) {
      v16h av, bv;
      if constexpr (AF16)
        av = frag_f16((const _Float16*)A + (size_t)(mi * 16 + lr) * lda + k0 + koff);
      else
        av = frag_f32((const float*)A + (size_t)(mi * 16 + lr) * lda + k0 + koff);
      if constexpr (BF16)
        bv = frag_f16((const _Float16*)B + (size_t)(ni * 16 + lr) * ldb + k0 + koff);
      else
        bv = frag_f32((const float*)B + (size_t)(ni * 16 + lr) * ldb + k0 + koff);
      acc = wmma32(av, bv, acc);
    }
    const int col   = ni * 16 + lr;
    const int rbase = mi * 16 + hl * 8;
    float sc = 1.f, sh = 0.f;
    if constexpr (EPI == 1) {
      float s = g[col] * rsqrtf(vv[col] + 1e-5f);
      sc = s; sh = bb[col] - mm[col] * s;
    }
#pragma unroll
    for (int r = 0; r < 8; ++r) {
      const int row = rbase + r;
      float x = acc[r];
      if constexpr (EPI == 1) { x = fmaf(x, sc, sh); x = fmaxf(x, 0.f); }
      if constexpr (EPI == 2) {
        float s = g[row] * rsqrtf(vv[row] + 1e-5f);
        x = fmaf(x, s, bb[row] - mm[row] * s);
        x = fmaxf(x, 0.f);
      }
      if constexpr (OUTF16)
        ((_Float16*)C)[(size_t)row * ldc + col] = (_Float16)x;
      else
        ((float*)C)[(size_t)row * ldc + col] = x;
    }
  }
}

// ---------------- K0: weight f32 -> f16 (once, then L2-resident) ----------
__global__ __launch_bounds__(256) void wcvt_kernel(const float* __restrict__ w1,
                                                   const float* __restrict__ w2,
                                                   _Float16* __restrict__ w1h,
                                                   _Float16* __restrict__ w2h) {
  const size_t i = (size_t)blockIdx.x * 256 + threadIdx.x;
  if (i < (size_t)256 * 4608) w1h[i] = (_Float16)w1[i];
  if (i < (size_t)128 * 256)  w2h[i] = (_Float16)w2[i];
}

// ---------------- K1: conv1 3x3 implicit GEMM + BN + leaky ----------------
// TILE_M = 128 pixels, N = all 256 out channels, K = 512*9 = 4608.
// 8 waves arranged 2(M) x 4(N); each wave owns a 64x64 register tile.
__global__ __launch_bounds__(256) void conv1_kernel(
    const float* __restrict__ x, const _Float16* __restrict__ w1h,
    const float* __restrict__ g1, const float* __restrict__ b1,
    const float* __restrict__ m1, const float* __restrict__ v1,
    _Float16* __restrict__ y) {
  constexpr int LDA = 40, LDB = 40;
  __shared__ _Float16 As[128 * LDA];   // im2col tile 128x32 (10.2 KB)
  __shared__ _Float16 Bs[256 * LDB];   // weight tile 256x32 (20.5 KB)

  const int tid = threadIdx.x, lane = tid & 31, wave = tid >> 5;
  const int hl = lane >> 4, lr = lane & 15, koff = hl ? 8 : 0;
  const int wave_m = wave >> 2, wave_n = wave & 3;
  const int m0 = blockIdx.x * 128;           // first pixel of tile
  const int n_img = m0 / 2304;               // 2304 % 128 == 0: no image straddle
  const int pix0 = m0 - n_img * 2304;
  const int kk = tid & 31;                   // staging column (loop-invariant)
  const int r0 = tid >> 5;                   // staging row start

  v8f zero = {};
  v8f acc[4][4];
#pragma unroll
  for (int mi = 0; mi < 4; ++mi)
#pragma unroll
    for (int ni = 0; ni < 4; ++ni) acc[mi][ni] = zero;

  for (int kt = 0; kt < 4608; kt += 32) {
    // ---- stage im2col A tile (hoisted k -> (c,kh,kw) decomposition) ----
    {
      const int k = kt + kk;
      const int c = k / 9;
      const int rem = k - c * 9;
      const int kh = rem / 3, kwv = rem - kh * 3;
      const float* xc = x + ((size_t)n_img * 512 + c) * 2304;
      for (int r = r0; r < 128; r += 8) {
        const int pix = pix0 + r;
        const int hh = pix / 48, ww = pix - hh * 48;
        const int ih = hh + kh - 1, iw = ww + kwv - 1;
        float val = 0.f;
        if ((unsigned)ih < 48u && (unsigned)iw < 48u) val = xc[ih * 48 + iw];
        As[r * LDA + kk] = (_Float16)val;
      }
    }
    // ---- stage weight tile (already f16) ----
    for (int r = r0; r < 256; r += 8)
      Bs[r * LDB + kk] = w1h[(size_t)r * 4608 + kt + kk];
    __syncthreads();

    v16h af[4];
#pragma unroll
    for (int mi = 0; mi < 4; ++mi)
      af[mi] = frag_f16(&As[(wave_m * 64 + mi * 16 + lr) * LDA + koff]);
#pragma unroll
    for (int ni = 0; ni < 4; ++ni) {
      v16h bf = frag_f16(&Bs[(wave_n * 64 + ni * 16 + lr) * LDB + koff]);
#pragma unroll
      for (int mi = 0; mi < 4; ++mi) acc[mi][ni] = wmma32(af[mi], bf, acc[mi][ni]);
    }
    __syncthreads();
  }

  // epilogue: BN + leaky-relu -> y (f16, [pix][256])
#pragma unroll
  for (int ni = 0; ni < 4; ++ni) {
    const int col = wave_n * 64 + ni * 16 + lr;
    const float s  = g1[col] * rsqrtf(v1[col] + 1e-5f);
    const float sh = b1[col] - m1[col] * s;
#pragma unroll
    for (int mi = 0; mi < 4; ++mi) {
      const int rb = wave_m * 64 + mi * 16 + hl * 8;
#pragma unroll
      for (int r = 0; r < 8; ++r) {
        float yv = fmaf(acc[mi][ni][r], s, sh);
        yv = yv > 0.f ? yv : 0.01f * yv;
        y[(size_t)(m0 + rb + r) * 256 + col] = (_Float16)yv;
      }
    }
  }
}

// ---------------- K2: conv2 1x1 (256->128) WMMA GEMM ----------------
// Block = 128 pixels x 128 channels, K = 256. One M-tile per wave; the full
// K-range of the A rows is held in registers (8 fragments), B streams from L2.
__global__ __launch_bounds__(256) void conv2_kernel(const _Float16* __restrict__ y,
                                                    const _Float16* __restrict__ w2h,
                                                    _Float16* __restrict__ x2) {
  const int tid = threadIdx.x, lane = tid & 31, wave = tid >> 5;
  const int hl = lane >> 4, lr = lane & 15, koff = hl ? 8 : 0;
  const int m0 = blockIdx.x * 128;
  const int arow = m0 + wave * 16 + lr;

  v16h afr[8];
#pragma unroll
  for (int k = 0; k < 8; ++k)
    afr[k] = frag_f16(y + (size_t)arow * 256 + k * 32 + koff);

#pragma unroll
  for (int ni = 0; ni < 8; ++ni) {
    v8f acc = {};
#pragma unroll
    for (int k = 0; k < 8; ++k) {
      v16h bf = frag_f16(w2h + (size_t)(ni * 16 + lr) * 256 + k * 32 + koff);
      acc = wmma32(afr[k], bf, acc);
    }
    const int col = ni * 16 + lr;
    const int rb = m0 + wave * 16 + hl * 8;
#pragma unroll
    for (int r = 0; r < 8; ++r)
      x2[(size_t)(rb + r) * 128 + col] = (_Float16)acc[r];
  }
}

// ---------------- K3: x_ave channel sums (deterministic) ----------------
__global__ __launch_bounds__(256) void xave_kernel(const _Float16* __restrict__ x2,
                                                   float* __restrict__ xave) {
  const int b = blockIdx.x;          // b = n*128 + c
  const int n = b >> 7, c = b & 127;
  __shared__ float red[256];
  float s = 0.f;
  for (int p = threadIdx.x; p < 2304; p += 256)
    s += (float)x2[((size_t)n * 2304 + p) * 128 + c];
  red[threadIdx.x] = s;
  __syncthreads();
  for (int st = 128; st > 0; st >>= 1) {
    if (threadIdx.x < st) red[threadIdx.x] += red[threadIdx.x + st];
    __syncthreads();
  }
  if (threadIdx.x == 0) xave[b] = red[0];
}

// ---------------- K4: cosine similarity map ----------------
__global__ __launch_bounds__(256) void cos_kernel(const _Float16* __restrict__ x2,
                                                  const float* __restrict__ xave,
                                                  float* __restrict__ cosg) {
  const int blk = blockIdx.x;          // 9 blocks per image
  const int n = blk / 9;
  const int tid = threadIdx.x;
  __shared__ float an[128];
  __shared__ float red[128];
  float av = 0.f;
  if (tid < 128) { av = xave[n * 128 + tid]; red[tid] = av * av; }
  __syncthreads();
  for (int st = 64; st > 0; st >>= 1) {
    if (tid < st) red[tid] += red[tid + st];
    __syncthreads();
  }
  if (tid < 128) an[tid] = av / fmaxf(sqrtf(red[0]), 1e-12f);
  __syncthreads();
  const size_t p = (size_t)blk * 256 + tid;
  const _Float16* row = x2 + p * 128;
  float dot = 0.f, nn = 0.f;
#pragma unroll 4
  for (int c = 0; c < 128; ++c) {
    float v = (float)row[c];
    dot = fmaf(an[c], v, dot);
    nn  = fmaf(v, v, nn);
  }
  cosg[p] = dot / fmaxf(sqrtf(nn), 1e-12f);
}

// ---------------- K5: per-image LBP + MLP + attention ----------------
__global__ __launch_bounds__(256) void lbp_attn_kernel(
    const float* __restrict__ cosg, const float* __restrict__ xave,
    const float* __restrict__ f1w,
    const float* __restrict__ f2w, const float* __restrict__ f2g,
    const float* __restrict__ f2b, const float* __restrict__ f2m,
    const float* __restrict__ f2v,
    const float* __restrict__ o1w, const float* __restrict__ o1g,
    const float* __restrict__ o1b, const float* __restrict__ o1m,
    const float* __restrict__ o1v,
    const float* __restrict__ kw, const float* __restrict__ qw,
    const float* __restrict__ vw,
    const float* __restrict__ ow, const float* __restrict__ og,
    const float* __restrict__ ob, const float* __restrict__ om,
    const float* __restrict__ ov,
    float* __restrict__ quantg, float* __restrict__ outs,
    char* __restrict__ img) {
  const int n = blockIdx.x, tid = threadIdx.x;
  __shared__ float sCos[2304];
  __shared__ float sCode[256];
  __shared__ float sRed[256];
  __shared__ float sQlv[128];
  __shared__ float sSta[128];

  char* base = img + (size_t)n * IMG_STRIDE;
  _Float16* h1g  = (_Float16*)(base + OFF_H1);
  _Float16* s_lm = (_Float16*)(base + OFF_S);
  _Float16* s2   = (_Float16*)(base + OFF_S2);
  _Float16* klm  = (_Float16*)(base + OFF_K);
  _Float16* qlm  = (_Float16*)(base + OFF_Q);
  _Float16* vcm  = (_Float16*)(base + OFF_V);
  float*    attF = (float*)   (base + OFF_AF);
  _Float16* a16  = (_Float16*)(base + OFF_A16);
  _Float16* flm  = (_Float16*)(base + OFF_F);
  _Float16* f2cm = (_Float16*)(base + OFF_F2);
  float* quantn = quantg + (size_t)n * 256 * 128;

  for (int i = tid; i < 2304; i += 256) sCos[i] = cosg[n * 2304 + i];
  __syncthreads();

  // LBP code per 16x16 fine cell position
  {
    const int lr = tid >> 4, lc = tid & 15;
    const float center = sCos[(16 + lr) * 48 + (16 + lc)];
    const float wts[9] = {1, 2, 4, 8, 0, 16, 32, 64, 128};
    float code = 0.f;
    int j = 0;
    for (int jr = 0; jr < 3; ++jr)
      for (int jc = 0; jc < 3; ++jc, ++j) {
        float cv = sCos[(jr * 16 + lr) * 48 + (jc * 16 + lc)];
        if (cv > center) code += wts[j];
      }
    sCode[tid] = code;
  }
  __syncthreads();
  // min / max of code
  sRed[tid] = sCode[tid]; __syncthreads();
  for (int st = 128; st > 0; st >>= 1) {
    if (tid < st) sRed[tid] = fminf(sRed[tid], sRed[tid + st]);
    __syncthreads();
  }
  const float mn = sRed[0]; __syncthreads();
  sRed[tid] = sCode[tid]; __syncthreads();
  for (int st = 128; st > 0; st >>= 1) {
    if (tid < st) sRed[tid] = fmaxf(sRed[tid], sRed[tid + st]);
    __syncthreads();
  }
  const float mx = sRed[0]; __syncthreads();
  const float codeN = (sCode[tid] - mn) / (mx - mn);
  sCode[tid] = codeN; __syncthreads();
  // cmin / cmax of normalized code
  sRed[tid] = codeN; __syncthreads();
  for (int st = 128; st > 0; st >>= 1) {
    if (tid < st) sRed[tid] = fminf(sRed[tid], sRed[tid + st]);
    __syncthreads();
  }
  const float cmin = sRed[0]; __syncthreads();
  sRed[tid] = codeN; __syncthreads();
  for (int st = 128; st > 0; st >>= 1) {
    if (tid < st) sRed[tid] = fmaxf(sRed[tid], sRed[tid + st]);
    __syncthreads();
  }
  const float cmax = sRed[0]; __syncthreads();

  if (tid < 128)
    sQlv[tid] = (2.f * tid + 1.f) * (1.f / 256.f) * (cmax - cmin) + cmin;
  __syncthreads();
  const float inter = sQlv[1] - sQlv[0];

  // quant (soft histogram membership) -> global scratch
  {
    const float cp = sCode[tid];
    const float thr = 1.f - inter;
    for (int q = 0; q < 128; ++q) {
      float d = 1.f - fabsf(sQlv[q] - cp);
      quantn[(size_t)tid * 128 + q] = (d > thr) ? d : 0.f;
    }
  }
  __threadfence();
  __syncthreads();
  // sta[q] = sum_p quant[p][q] (deterministic per-thread sum), then normalize
  if (tid < 128) {
    float s = 0.f;
    for (int p = 0; p < 256; ++p) s += quantn[(size_t)p * 128 + tid];
    sSta[tid] = s;
    sRed[tid] = s;
  }
  __syncthreads();
  for (int st = 64; st > 0; st >>= 1) {
    if (tid < st) sRed[tid] += sRed[tid + st];
    __syncthreads();
  }
  if (tid < 128) sSta[tid] = sSta[tid] / sRed[0];
  __syncthreads();

  // h1 = leaky(f1_w (64x2) @ [qlv; sta]) stored level-major 128x64 f16
  for (int idx = tid; idx < 128 * 64; idx += 256) {
    const int l = idx >> 6, o = idx & 63;
    float hv = f1w[o * 2 + 0] * sQlv[l] + f1w[o * 2 + 1] * sSta[l];
    hv = hv > 0.f ? hv : 0.01f * hv;
    h1g[(size_t)l * 64 + o] = (_Float16)hv;
  }
  // xa: broadcast x_ave into s cols 128..255
  for (int idx = tid; idx < 128 * 128; idx += 256) {
    const int l = idx >> 7, c = idx & 127;
    s_lm[(size_t)l * 256 + 128 + c] = (_Float16)(xave[n * 128 + c] * (1.f / 2304.f));
  }
  __threadfence(); __syncthreads();

  // h2 = relu(bn(f2_w @ h1)) -> s cols 0..127
  gemm_block<true, false, 1, true>(h1g, 64, f2w, 64, s_lm, 256, 128, 128, 64,
                                   f2g, f2b, f2m, f2v);
  __threadfence(); __syncthreads();
  // s2 = relu(bn(out1_w @ s))
  gemm_block<true, false, 1, true>(s_lm, 256, o1w, 256, s2, 256, 128, 256, 256,
                                   o1g, o1b, o1m, o1v);
  __threadfence(); __syncthreads();
  // k, q level-major; v channel-major
  gemm_block<true, false, 0, true>(s2, 256, kw, 256, klm, 256, 128, 256, 256,
                                   nullptr, nullptr, nullptr, nullptr);
  gemm_block<true, false, 0, true>(s2, 256, qw, 256, qlm, 256, 128, 256, 256,
                                   nullptr, nullptr, nullptr, nullptr);
  gemm_block<false, true, 0, true>(vw, 256, s2, 256, vcm, 128, 256, 128, 256,
                                   nullptr, nullptr, nullptr, nullptr);
  __threadfence(); __syncthreads();
  // attn[l][m] = sum_c k[l][c] q[m][c]
  gemm_block<true, true, 0, false>(klm, 256, qlm, 256, attF, 128, 128, 128, 256,
                                   nullptr, nullptr, nullptr, nullptr);
  __threadfence(); __syncthreads();
  // softmax over m, store f16
  if (tid < 128) {
    const float* row = attF + (size_t)tid * 128;
    float rmx = -3.402823466e38f;
    for (int m = 0; m < 128; ++m) rmx = fmaxf(rmx, row[m]);
    float ssum = 0.f;
    for (int m = 0; m < 128; ++m) ssum += __expf(row[m] - rmx);
    const float inv = 1.f / ssum;
    for (int m = 0; m < 128; ++m)
      a16[(size_t)tid * 128 + m] = (_Float16)(__expf(row[m] - rmx) * inv);
  }
  __threadfence(); __syncthreads();
  // f[l][c] = sum_m attn[l][m] v[c][m]
  gemm_block<true, true, 0, true>(a16, 128, vcm, 128, flm, 256, 128, 256, 128,
                                  nullptr, nullptr, nullptr, nullptr);
  __threadfence(); __syncthreads();
  // f2[c][l] = relu(bn(out_w @ f)) channel-major
  gemm_block<false, true, 2, true>(ow, 256, flm, 256, f2cm, 128, 256, 128, 256,
                                   og, ob, om, ov);
  __threadfence(); __syncthreads();
  // out[c][p] = sum_l f2[c][l] quant[p][l]
  gemm_block<true, false, 0, false>(f2cm, 128, quantn, 128,
                                    outs + (size_t)n * 65536, 256,
                                    256, 256, 128,
                                    nullptr, nullptr, nullptr, nullptr);
}

// ---------------- K6: bilinear resize 16x16 -> 48x48 ----------------
__global__ __launch_bounds__(256) void resize_kernel(const float* __restrict__ outs,
                                                     float* __restrict__ out) {
  const size_t idx = (size_t)blockIdx.x * 256 + threadIdx.x;
  const size_t total = (size_t)32 * 256 * 48 * 48;
  if (idx >= total) return;
  const int wv = (int)(idx % 48);
  const int hv = (int)((idx / 48) % 48);
  const size_t nc = idx / 2304;
  const float* s = outs + nc * 256;
  const float fy = hv * (15.f / 47.f);
  const float fx = wv * (15.f / 47.f);
  const int y0 = (int)floorf(fy);
  const int x0 = (int)floorf(fx);
  const int y1 = y0 + 1 < 16 ? y0 + 1 : 15;
  const int x1 = x0 + 1 < 16 ? x0 + 1 : 15;
  const float wy = fy - (float)y0, wx = fx - (float)x0;
  const float v00 = s[y0 * 16 + x0], v01 = s[y0 * 16 + x1];
  const float v10 = s[y1 * 16 + x0], v11 = s[y1 * 16 + x1];
  out[idx] = v00 * (1.f - wy) * (1.f - wx) + v01 * (1.f - wy) * wx +
             v10 * wy * (1.f - wx) + v11 * wy * wx;
}

// ---------------- host launcher ----------------
extern "C" void kernel_launch(void* const* d_in, const int* in_sizes, int n_in,
                              void* d_out, int out_size, void* d_ws, size_t ws_size,
                              hipStream_t stream) {
  (void)in_sizes; (void)n_in; (void)out_size; (void)ws_size;
  const float* x   = (const float*)d_in[0];
  const float* w1  = (const float*)d_in[1];
  const float* g1  = (const float*)d_in[2];
  const float* b1  = (const float*)d_in[3];
  const float* m1  = (const float*)d_in[4];
  const float* v1  = (const float*)d_in[5];
  const float* w2  = (const float*)d_in[6];
  const float* f1w = (const float*)d_in[7];
  const float* f2w = (const float*)d_in[8];
  const float* f2g = (const float*)d_in[9];
  const float* f2b = (const float*)d_in[10];
  const float* f2m = (const float*)d_in[11];
  const float* f2v = (const float*)d_in[12];
  const float* o1w = (const float*)d_in[13];
  const float* o1g = (const float*)d_in[14];
  const float* o1b = (const float*)d_in[15];
  const float* o1m = (const float*)d_in[16];
  const float* o1v = (const float*)d_in[17];
  const float* kw  = (const float*)d_in[18];
  const float* qw  = (const float*)d_in[19];
  const float* vw  = (const float*)d_in[20];
  const float* ow  = (const float*)d_in[21];
  const float* og  = (const float*)d_in[22];
  const float* ob  = (const float*)d_in[23];
  const float* om  = (const float*)d_in[24];
  const float* ov  = (const float*)d_in[25];

  char* ws = (char*)d_ws;
  _Float16* X2   = (_Float16*)(ws + X2_OFF);
  float*    XAVE = (float*)(ws + XAVE_OFF);
  float*    COSB = (float*)(ws + COS_OFF);
  float*    QNT  = (float*)(ws + QUANT_OFF);
  float*    OUTS = (float*)(ws + OUTS_OFF);
  char*     IMG  = ws + IMG_OFF;
  _Float16* W1H  = (_Float16*)(ws + W1H_OFF);
  _Float16* W2H  = (_Float16*)(ws + W2H_OFF);
  _Float16* Y    = (_Float16*)(ws + Y_OFF);

  wcvt_kernel<<<4608, 256, 0, stream>>>(w1, w2, W1H, W2H);
  conv1_kernel<<<576, 256, 0, stream>>>(x, W1H, g1, b1, m1, v1, Y);
  conv2_kernel<<<576, 256, 0, stream>>>(Y, W2H, X2);
  xave_kernel<<<4096, 256, 0, stream>>>(X2, XAVE);
  cos_kernel<<<288, 256, 0, stream>>>(X2, XAVE, COSB);
  lbp_attn_kernel<<<32, 256, 0, stream>>>(COSB, XAVE, f1w,
                                          f2w, f2g, f2b, f2m, f2v,
                                          o1w, o1g, o1b, o1m, o1v,
                                          kw, qw, vw,
                                          ow, og, ob, om, ov,
                                          QNT, OUTS, IMG);
  resize_kernel<<<73728, 256, 0, stream>>>(OUTS, (float*)d_out);
}
